// MyYolo_32598801777047
// MI455X (gfx1250) — compile-verified, matrix operand order checked by reading, not run on hardware
//
#include <hip/hip_runtime.h>
#include <hip/hip_bf16.h>
#include <math.h>
#include <stdint.h>

typedef _Float16 half_t;
typedef __attribute__((ext_vector_type(16))) _Float16 v16h;
typedef __attribute__((ext_vector_type(8)))  float    v8f;

#define ACT_NONE    0
#define ACT_RELU    1
#define ACT_LEAKY   2
#define ACT_ADDRELU 3
#define ACT_SAM     4

// ---------------------------------------------------------------------------
// fp32 -> fp16 convert
// ---------------------------------------------------------------------------
__global__ void yolo_f32_to_f16(const float* __restrict__ src, half_t* __restrict__ dst, int n) {
    int i = blockIdx.x * blockDim.x + threadIdx.x;
    if (i < n) dst[i] = (half_t)src[i];
}

// ---------------------------------------------------------------------------
// Pack OIHW fp32 weights into WMMA B-fragment order (fp16).
// Fragment layout: dst[((kc*NT + nt)*32 + lane)*16 + j]
//   k = kc*32 + (lane>>4)*8 + (j<8 ? j : j+8)      (mirrors 16-bit A layout)
//   n = nt*16 + (lane&15)
// ---------------------------------------------------------------------------
__global__ void yolo_pack_w(const float* __restrict__ w, half_t* __restrict__ dst,
                            int cout, int cin, int ksz, int NT, int KC) {
    long tid = (long)blockIdx.x * blockDim.x + threadIdx.x;
    long total = (long)KC * NT * 512;
    if (tid >= total) return;
    int j    = (int)(tid & 15);
    int lane = (int)((tid >> 4) & 31);
    int frag = (int)(tid >> 9);
    int nt   = frag % NT;
    int kc   = frag / NT;
    int n    = nt * 16 + (lane & 15);
    int k    = kc * 32 + (lane >> 4) * 8 + (j < 8 ? j : j + 8);
    int khw  = ksz * ksz;
    int K    = cin * khw;
    float v  = 0.f;
    if (n < cout && k < K) {
        int c = k / khw; int rs = k - c * khw;
        int r = rs / ksz; int s = rs - r * ksz;
        v = w[(((long)n * cin + c) * ksz + r) * ksz + s];
    }
    dst[tid] = (half_t)v;
}

// ---------------------------------------------------------------------------
// Implicit-GEMM convolution via V_WMMA_F32_16X16X32_F16.
// Workgroup = 4 waves; wave w computes pixel tile (blockIdx.x*4 + w) x the
// shared 64-cout panel nt0..nt0+3. The B panel (4 KB per K-chunk) is staged
// global -> LDS with GLOBAL_LOAD_ASYNC_TO_LDS_B128 (ASYNCcnt), double
// buffered so the next chunk's DMA overlaps the current chunk's WMMAs, and
// consumed with ds_load_b128. A fragments are gathered per lane; KSZ is a
// template parameter so (c,r,s) decomposition uses constant-divisor mults,
// and KSZ==1 is a divide-free, check-free gather.
// Fused epilogue: BN scale/bias, relu / leaky / residual-add-relu / SAM.
// ---------------------------------------------------------------------------
template<int KSZ>
__global__ __launch_bounds__(128) void yolo_conv_wmma(
    const half_t* __restrict__ in, const half_t* __restrict__ wp,
    const float* __restrict__ bns, const float* __restrict__ bnb,
    half_t* __restrict__ out, const half_t* __restrict__ res, float* __restrict__ out32,
    int cin, int cout, int Hin, int Win, int Hout, int Wout,
    int stride, int pad, int act, int M, int NT, int KC)
{
    constexpr int KHW = KSZ * KSZ;
    __shared__ half_t sB[2][2048];              // 2 x 4KB B-panel double buffer

    const int tid     = threadIdx.x;
    const int lane    = tid & 31;
    const int wave    = tid >> 5;
    const int halfsel = lane >> 4;
    const int lan15   = lane & 15;
    const int m_base  = (blockIdx.x * 4 + wave) * 16;
    const int nt0     = blockIdx.y * 4;

    const int m      = m_base + lan15;          // A-fragment row for this lane
    const bool mval  = (m < M);
    const int oy     = mval ? (m / Wout) : 0;
    const int ox     = mval ? (m - oy * Wout) : 0;
    const int K      = cin * KHW;
    const int Keff   = mval ? K : 0;            // folds the m-valid predicate
    const int iy0    = oy * stride - pad;
    const int ix0    = ox * stride - pad;
    const long HW    = (long)Hin * Win;
    const long base0 = (long)iy0 * Win + ix0;   // KSZ==1 fast path (pad==0)

    const uint32_t lds0 = (uint32_t)(uintptr_t)&sB[0][0];   // LDS byte offset

    // Cooperative async copy of one 4KB B panel (K-chunk kc) into sB[buf].
    // 256 x 16B chunks; each of the 128 threads issues 2 async b128 ops.
    auto issueB = [&](int kc, int buf) {
        const half_t* src = wp + ((long)kc * NT + nt0) * 512;
        #pragma unroll
        for (int c = 0; c < 2; ++c) {
            int chunk = tid + c * 128;
            uint32_t laddr = lds0 + (uint32_t)buf * 4096u + (uint32_t)chunk * 16u;
            unsigned long long gaddr = (unsigned long long)(uintptr_t)(src + chunk * 8);
            asm volatile("global_load_async_to_lds_b128 %0, %1, off"
                         :: "v"(laddr), "v"(gaddr) : "memory");
        }
    };

    v8f acc0 = {}; v8f acc1 = {}; v8f acc2 = {}; v8f acc3 = {};

    issueB(0, 0);
    for (int kc = 0; kc < KC; ++kc) {
        const int buf = kc & 1;
        if (kc + 1 < KC) {
            issueB(kc + 1, buf ^ 1);            // overlaps with this chunk's WMMAs
            asm volatile("s_wait_asynccnt 0x2" ::: "memory");   // chunk kc landed
        } else {
            asm volatile("s_wait_asynccnt 0x0" ::: "memory");
        }
        __syncthreads();                        // panel kc visible to all waves

        // ---- gather A fragment (16x32 f16, documented CDNA5 layout) ----
        v16h a;
        const int kb = kc * 32 + halfsel * 8;
        #pragma unroll
        for (int j = 0; j < 16; ++j) {
            const int k = kb + (j < 8 ? j : j + 8);
            half_t v = (half_t)0.f;
            if constexpr (KSZ == 1) {
                if (k < Keff) v = in[(long)k * HW + base0];
            } else {
                if (k < Keff) {
                    int c  = k / KHW;               // constant divisor
                    int rs = k - c * KHW;
                    int r  = rs / KSZ;              // constant divisor
                    int s  = rs - r * KSZ;
                    int iy = iy0 + r;
                    int ix = ix0 + s;
                    if (iy >= 0 && iy < Hin && ix >= 0 && ix < Win)
                        v = in[c * HW + (long)iy * Win + ix];
                }
            }
            a[j] = v;
        }

        // ---- B fragments from LDS: tile t at sB[buf][(t*32+lane)*16] ----
        const v16h* bp = (const v16h*)&sB[buf][(long)lane * 16];
        if (nt0 + 0 < NT) { v16h b = bp[0];
            acc0 = __builtin_amdgcn_wmma_f32_16x16x32_f16(false, a, false, b, (short)0, acc0, false, false); }
        if (nt0 + 1 < NT) { v16h b = bp[32];
            acc1 = __builtin_amdgcn_wmma_f32_16x16x32_f16(false, a, false, b, (short)0, acc1, false, false); }
        if (nt0 + 2 < NT) { v16h b = bp[64];
            acc2 = __builtin_amdgcn_wmma_f32_16x16x32_f16(false, a, false, b, (short)0, acc2, false, false); }
        if (nt0 + 3 < NT) { v16h b = bp[96];
            acc3 = __builtin_amdgcn_wmma_f32_16x16x32_f16(false, a, false, b, (short)0, acc3, false, false); }

        __syncthreads();                        // everyone done reading sB[buf]
    }

    // ---- epilogue: C/D layout = { row r+8*halfsel, col lane&15 } ----
    v8f accs[4] = {acc0, acc1, acc2, acc3};
    #pragma unroll
    for (int t = 0; t < 4; ++t) {
        const int nt = nt0 + t;
        if (nt >= NT) break;
        const int n = nt * 16 + lan15;
        if (n >= cout) continue;
        const float sc = bns ? bns[n] : 1.f;
        const float bi = bnb ? bnb[n] : 0.f;
        #pragma unroll
        for (int r = 0; r < 8; ++r) {
            const int mm = m_base + r + halfsel * 8;
            if (mm >= M) continue;
            float v = accs[t][r] * sc + bi;
            if      (act == ACT_RELU)    v = v > 0.f ? v : 0.f;
            else if (act == ACT_LEAKY)   v = v > 0.f ? v : 0.1f * v;
            else if (act == ACT_ADDRELU) { v += (float)res[(long)n * M + mm]; v = v > 0.f ? v : 0.f; }
            else if (act == ACT_SAM)     { float x = (float)res[(long)n * M + mm];
                                           v = x * (1.f / (1.f + __expf(-v))); }
            if (out)   out[(long)n * M + mm]   = (half_t)v;
            if (out32) out32[(long)n * M + mm] = v;
        }
    }
}

// ---------------------------------------------------------------------------
// Max pooling (fp16 CHW). k==1,s==1,p==0 doubles as a copy (SPP concat).
// ---------------------------------------------------------------------------
__global__ void yolo_maxpool(const half_t* __restrict__ in, half_t* __restrict__ out,
                             int C, int Hin, int Win, int Hout, int Wout,
                             int k, int stride, int pad) {
    long tid = (long)blockIdx.x * blockDim.x + threadIdx.x;
    long total = (long)C * Hout * Wout;
    if (tid >= total) return;
    int hw = Hout * Wout;
    int c  = (int)(tid / hw);
    int mm = (int)(tid - (long)c * hw);
    int oy = mm / Wout, ox = mm - oy * Wout;
    float best = -INFINITY;
    for (int r = 0; r < k; ++r) {
        int iy = oy * stride + r - pad;
        if (iy < 0 || iy >= Hin) continue;
        for (int s = 0; s < k; ++s) {
            int ix = ox * stride + s - pad;
            if (ix < 0 || ix >= Win) continue;
            float v = (float)in[((long)c * Hin + iy) * Win + ix];
            best = fmaxf(best, v);
        }
    }
    out[tid] = (half_t)best;
}

// ---------------------------------------------------------------------------
// Decode: pred [85][400] fp32 -> boxes/scores/cls_id in d_out
// ---------------------------------------------------------------------------
__global__ void yolo_decode(const float* __restrict__ pred, float* __restrict__ o) {
    int i = blockIdx.x * blockDim.x + threadIdx.x;
    if (i >= 400) return;
    const int M = 400;
    float conf = 1.f / (1.f + __expf(-pred[0 * M + i]));
    float mx = -INFINITY; int arg = 0;
    for (int c = 0; c < 80; ++c) {
        float l = pred[(1 + c) * M + i];
        if (l > mx) { mx = l; arg = c; }
    }
    float sum = 0.f;
    for (int c = 0; c < 80; ++c) sum += __expf(pred[(1 + c) * M + i] - mx);
    float score = conf / sum;                         // softmax(argmax)*conf
    float gx = (float)(i % 20), gy = (float)(i / 20);
    float cx = (1.f / (1.f + __expf(-pred[81 * M + i])) + gx) * 32.f;
    float cy = (1.f / (1.f + __expf(-pred[82 * M + i])) + gy) * 32.f;
    float w  = __expf(pred[83 * M + i]);
    float h  = __expf(pred[84 * M + i]);
    const float inv = 1.f / 640.f;
    o[i * 4 + 0] = fminf(fmaxf((cx - 0.5f * w) * inv, 0.f), 1.f);
    o[i * 4 + 1] = fminf(fmaxf((cy - 0.5f * h) * inv, 0.f), 1.f);
    o[i * 4 + 2] = fminf(fmaxf((cx + 0.5f * w) * inv, 0.f), 1.f);
    o[i * 4 + 3] = fminf(fmaxf((cy + 0.5f * h) * inv, 0.f), 1.f);
    o[1600 + i] = score;
    o[2000 + i] = (float)arg;
}

// ---------------------------------------------------------------------------
// NMS, single block. Stable rank-based argsort(-scores) + sequential greedy,
// replicating the reference's IoU formula exactly.
// ---------------------------------------------------------------------------
__global__ __launch_bounds__(512) void yolo_nms(float* __restrict__ io) {
    const int N = 400;
    __shared__ float sx1[N], sy1[N], sx2[N], sy2[N], ss[N], sa[N];
    __shared__ int   scls[N], order[N];
    __shared__ unsigned char keep[N];
    __shared__ int flag;
    int t = threadIdx.x;
    for (int i = t; i < N; i += blockDim.x) {
        sx1[i] = io[i * 4 + 0]; sy1[i] = io[i * 4 + 1];
        sx2[i] = io[i * 4 + 2]; sy2[i] = io[i * 4 + 3];
        ss[i]  = io[1600 + i];  scls[i] = (int)io[2000 + i];
        sa[i]  = (sx2[i] - sx1[i]) * (sy2[i] - sy1[i]);
    }
    __syncthreads();
    for (int i = t; i < N; i += blockDim.x) {
        float si = ss[i]; int rank = 0;
        for (int j = 0; j < N; ++j) {
            float sj = ss[j];
            if (sj > si || (sj == si && j < i)) rank++;
        }
        order[rank] = i;
    }
    __syncthreads();
    for (int i = 0; i < N; ++i) {
        if (t == 0) flag = 0;
        __syncthreads();
        int oi = order[i];
        for (int jj = t; jj < i; jj += blockDim.x) {
            if (keep[jj] && scls[order[jj]] == scls[oi]) {
                int oj = order[jj];
                float iw = fmaxf(1e-28f, fmaxf(sx2[oi], sx2[oj]) - fmaxf(sx1[oi], sx1[oj]));
                float ih = fmaxf(1e-28f, fmaxf(sy2[oi], sy2[oj]) - fmaxf(sy1[oi], sy1[oj]));
                float inter = iw * ih;
                float iou = inter / (sa[oi] + sa[oj] - inter);
                if (iou > 0.5f) flag = 1;
            }
        }
        __syncthreads();
        if (t == 0) keep[i] = (ss[oi] > 0.01f) && !flag;
        __syncthreads();
    }
    for (int i = t; i < N; i += blockDim.x) io[2400 + order[i]] = keep[i] ? 1.f : 0.f;
}

// ---------------------------------------------------------------------------
// Host orchestration
// ---------------------------------------------------------------------------
struct CD { int wi, si, bi, cout, cin, k; };

extern "C" void kernel_launch(void* const* d_in, const int* in_sizes, int n_in,
                              void* d_out, int out_size, void* d_ws, size_t ws_size,
                              hipStream_t stream) {
    (void)in_sizes; (void)n_in; (void)out_size; (void)ws_size;

    // 27 conv layers in setup_inputs() flattening order.
    static const CD cds[27] = {
        { 1, 2, 3,  64,   3, 7},                                       // conv1
        { 4, 5, 6,  64,  64, 3}, { 7, 8, 9,  64,  64, 3},              // l1b0
        {10,11,12,  64,  64, 3}, {13,14,15,  64,  64, 3},              // l1b1
        {16,17,18, 128,  64, 3}, {19,20,21, 128, 128, 3}, {22,23,24, 128,  64, 1}, // l2b0 (c1,c2,d)
        {25,26,27, 128, 128, 3}, {28,29,30, 128, 128, 3},              // l2b1
        {31,32,33, 256, 128, 3}, {34,35,36, 256, 256, 3}, {37,38,39, 256, 128, 1}, // l3b0
        {40,41,42, 256, 256, 3}, {43,44,45, 256, 256, 3},              // l3b1
        {46,47,48, 512, 256, 3}, {49,50,51, 512, 512, 3}, {52,53,54, 512, 256, 1}, // l4b0
        {55,56,57, 512, 512, 3}, {58,59,60, 512, 512, 3},              // l4b1
        {61,62,63, 512,2048, 1},                                       // spp
        {64,-1,65, 512, 512, 1},                                       // sam (bias only)
        {66,67,68, 256, 512, 1}, {69,70,71, 512, 256, 3},              // cs0, cs1
        {72,73,74, 256, 512, 1}, {75,76,77, 512, 256, 3},              // cs2, cs3
        {78,-1,79,  85, 512, 1},                                       // pred (bias only)
    };

    // ---- workspace carve-up (bytes) ----
    char* ws = (char*)d_ws;
    half_t* wp   = (half_t*)(ws + 0);                 // packed weights (~30 MB)
    half_t* img  = (half_t*)(ws + (40l << 20));       // fp16 image 3x640x640
    half_t* A0   = (half_t*)(ws + (43l << 20));       // activation buffers (14 MB each)
    half_t* A1   = (half_t*)(ws + (57l << 20));
    half_t* A2   = (half_t*)(ws + (71l << 20));
    half_t* SPP  = (half_t*)(ws + (85l << 20));       // 2048 x 400 fp16
    float*  PRED = (float*) (ws + (87l << 20));       // 85 x 400 fp32

    // ---- pack all weights to fp16 fragment order ----
    size_t offs[27]; size_t off = 0;
    for (int i = 0; i < 27; ++i) {
        const CD& d = cds[i];
        int KC = (d.cin * d.k * d.k + 31) / 32;
        int NT = (d.cout + 15) / 16;
        offs[i] = off;
        long total = (long)KC * NT * 512;
        yolo_pack_w<<<dim3((unsigned)((total + 255) / 256)), 256, 0, stream>>>(
            (const float*)d_in[d.wi], wp + off, d.cout, d.cin, d.k, NT, KC);
        off += (size_t)total;
    }

    auto conv = [&](const half_t* inp, int Hin, int Win, int di,
                    int stride, int pad, int Hout, int Wout,
                    half_t* outp, int act, const half_t* resp, float* out32p) {
        const CD& d = cds[di];
        int M  = Hout * Wout;
        int NT = (d.cout + 15) / 16;
        int KC = (d.cin * d.k * d.k + 31) / 32;
        dim3 g((unsigned)((M + 63) / 64), (unsigned)((NT + 3) / 4));
        const float* s = d.si >= 0 ? (const float*)d_in[d.si] : nullptr;
        const float* b = d.bi >= 0 ? (const float*)d_in[d.bi] : nullptr;
        if (d.k == 1)
            yolo_conv_wmma<1><<<g, 128, 0, stream>>>(inp, wp + offs[di], s, b, outp, resp, out32p,
                d.cin, d.cout, Hin, Win, Hout, Wout, stride, pad, act, M, NT, KC);
        else if (d.k == 3)
            yolo_conv_wmma<3><<<g, 128, 0, stream>>>(inp, wp + offs[di], s, b, outp, resp, out32p,
                d.cin, d.cout, Hin, Win, Hout, Wout, stride, pad, act, M, NT, KC);
        else
            yolo_conv_wmma<7><<<g, 128, 0, stream>>>(inp, wp + offs[di], s, b, outp, resp, out32p,
                d.cin, d.cout, Hin, Win, Hout, Wout, stride, pad, act, M, NT, KC);
    };

    // ---- stem ----
    int nimg = 3 * 640 * 640;
    yolo_f32_to_f16<<<(nimg + 255) / 256, 256, 0, stream>>>((const float*)d_in[0], img, nimg);
    conv(img, 640, 640, 0, 2, 3, 320, 320, A0, ACT_RELU, nullptr, nullptr);      // conv1 -> 64x320x320
    {
        long total = 64l * 160 * 160;
        yolo_maxpool<<<dim3((unsigned)((total + 255) / 256)), 256, 0, stream>>>(
            A0, A1, 64, 320, 320, 160, 160, 3, 2, 1);                            // -> 64x160x160
    }

    // ---- residual stages ----
    half_t* X = A1; half_t* T = A0; half_t* U = A2;
    int H = 160, W = 160;
    int li = 1;
    for (int l = 0; l < 4; ++l) {
        for (int b = 0; b < 2; ++b) {
            bool hasd  = (l > 0 && b == 0);
            int stride = hasd ? 2 : 1;
            int Ho = H / stride, Wo = W / stride;
            conv(X, H, W, li, stride, 1, Ho, Wo, T, ACT_RELU, nullptr, nullptr);      // c1
            half_t* resbuf; half_t* outbuf;
            if (hasd) {
                conv(X, H, W, li + 2, stride, 0, Ho, Wo, U, ACT_NONE, nullptr, nullptr); // downsample
                resbuf = U; outbuf = X;
            } else { resbuf = X; outbuf = U; }
            conv(T, Ho, Wo, li + 1, 1, 1, Ho, Wo, outbuf, ACT_ADDRELU, resbuf, nullptr); // c2 + idn
            li += hasd ? 3 : 2;
            if (!hasd) { half_t* tmp = X; X = U; U = tmp; }
            H = Ho; W = Wo;
        }
    }
    // X = 512 x 20 x 20

    // ---- SPP concat [x, mp5, mp9, mp13] -> 2048 x 400 ----
    {
        long total = 512l * 400;
        unsigned gb = (unsigned)((total + 255) / 256);
        yolo_maxpool<<<gb, 256, 0, stream>>>(X, SPP,               512, 20, 20, 20, 20,  1, 1, 0);
        yolo_maxpool<<<gb, 256, 0, stream>>>(X, SPP + 512l * 400,  512, 20, 20, 20, 20,  5, 1, 2);
        yolo_maxpool<<<gb, 256, 0, stream>>>(X, SPP + 1024l * 400, 512, 20, 20, 20, 20,  9, 1, 4);
        yolo_maxpool<<<gb, 256, 0, stream>>>(X, SPP + 1536l * 400, 512, 20, 20, 20, 20, 13, 1, 6);
    }
    conv(SPP, 20, 20, 20, 1, 0, 20, 20, T, ACT_LEAKY, nullptr, nullptr);   // spp 2048->512 in T

    // ---- SAM: out = x * sigmoid(conv1x1(x) + b) ----
    conv(T, 20, 20, 21, 1, 0, 20, 20, U, ACT_SAM, T, nullptr);             // -> U (512)

    // ---- head ----
    conv(U, 20, 20, 22, 1, 0, 20, 20, T, ACT_LEAKY, nullptr, nullptr);     // cs0 512->256
    conv(T, 20, 20, 23, 1, 1, 20, 20, U, ACT_LEAKY, nullptr, nullptr);     // cs1 256->512
    conv(U, 20, 20, 24, 1, 0, 20, 20, T, ACT_LEAKY, nullptr, nullptr);     // cs2 512->256
    conv(T, 20, 20, 25, 1, 1, 20, 20, U, ACT_LEAKY, nullptr, nullptr);     // cs3 256->512
    conv(U, 20, 20, 26, 1, 0, 20, 20, nullptr, ACT_NONE, nullptr, PRED);   // pred 512->85 fp32

    // ---- decode + NMS ----
    yolo_decode<<<1, 512, 0, stream>>>(PRED, (float*)d_out);
    yolo_nms<<<1, 512, 0, stream>>>((float*)d_out);
}
